// TimeMix_23605140259524
// MI455X (gfx1250) — matrix operand order repaired and checked by reading, not compiled, not run on hardware
//
#include <hip/hip_runtime.h>

typedef __bf16 bf16;
typedef __attribute__((ext_vector_type(16))) __bf16 v16bf;
typedef __attribute__((ext_vector_type(8)))  __bf16 v8bf;
typedef __attribute__((ext_vector_type(8)))  float  v8f;

#define B_  4
#define S_  4096
#define D_  2048
#define M_  (B_ * S_)      // 16384 rows
#define K_  D_             // 2048
#define N_  D_             // 2048
#define NC  32             // scan chunks
#define CL  (S_ / NC)      // 128 steps per chunk

// ---------------------------------------------------------------- fp32 -> bf16
__global__ __launch_bounds__(256) void cvt_kernel(const float* __restrict__ src,
                                                  bf16* __restrict__ dst, int n) {
    int i = blockIdx.x * blockDim.x + threadIdx.x;
    if (i < n) dst[i] = (bf16)src[i];
}

// -------------------------------------------------- WMMA tile loaders (wave32)
// A: 16x32 bf16 tile of X (row-major, ld=K_). Per ISA: lane<16 -> row=lane,
// K = {0..7, 16..23}; lane>=16 -> row=lane-16, K = {8..15, 24..31}.
__device__ __forceinline__ v16bf load_A(const bf16* __restrict__ X,
                                        int m0, int k0, int lane) {
    int hf = lane >> 4, l = lane & 15;
    const bf16* p = X + (size_t)(m0 + l) * K_ + k0 + hf * 8;
    v8bf lo = *(const v8bf*)(p);
    v8bf hi = *(const v8bf*)(p + 16);
    v16bf a;
#pragma unroll
    for (int i = 0; i < 8; ++i) { a[i] = lo[i]; a[8 + i] = hi[i]; }
    return a;
}

// B: 32x16 bf16 tile where B[k][n] = W[n][k], W row-major [N_][K_].
// lane<16 -> col n=lane, K = {0..7, 16..23}; lane>=16 -> K = {8..15, 24..31}.
__device__ __forceinline__ v16bf load_B(const bf16* __restrict__ W,
                                        int n0, int k0, int lane) {
    int hf = lane >> 4, l = lane & 15;
    const bf16* p = W + (size_t)(n0 + l) * K_ + k0 + hf * 8;
    v8bf lo = *(const v8bf*)(p);
    v8bf hi = *(const v8bf*)(p + 16);
    v16bf b;
#pragma unroll
    for (int i = 0; i < 8; ++i) { b[i] = lo[i]; b[8 + i] = hi[i]; }
    return b;
}

// ---------------------------------------------- epilogue for one 16x16 tile
__device__ __forceinline__ void proj_epilogue(int m0, int n0, int lane,
    const v8f& aR, const v8f& aW, const v8f& aK, const v8f& aV,
    bf16* __restrict__ r_out, float* __restrict__ ewk_out, bf16* __restrict__ v_out)
{
    int hf = lane >> 4, l = lane & 15;
#pragma unroll
    for (int i = 0; i < 8; ++i) {
        size_t idx = (size_t)(m0 + i + 8 * hf) * N_ + (n0 + l);
        float rr = 1.0f / (1.0f + __expf(-aR[i]));      // sigmoid
        float ww = -__expf(aW[i]);                      // w = -exp(xWw)
        float ek = __expf(ww + aK[i]);                  // exp(w + k)
        r_out[idx]   = (bf16)rr;
        ewk_out[idx] = ek;
        v_out[idx]   = (bf16)aV[i];
    }
}

// -------------------------------- fused r/w/k/v projections, 32x16 per wave
__global__ __launch_bounds__(256) void proj_kernel(
    const bf16* __restrict__ Xb,  const bf16* __restrict__ Wrb,
    const bf16* __restrict__ Wwb, const bf16* __restrict__ Wkb,
    const bf16* __restrict__ Wvb,
    bf16* __restrict__ r_out, float* __restrict__ ewk_out, bf16* __restrict__ v_out)
{
    int lane = threadIdx.x & 31;
    int wave = blockIdx.x * (blockDim.x >> 5) + (threadIdx.x >> 5);
    int tn = wave & (N_ / 16 - 1);        // 0..127
    int tp = wave / (N_ / 16);            // pair of M-tiles, 0..511
    if (tp >= M_ / 32) return;            // wave-uniform
    int m0 = tp * 32, n0 = tn * 16;

    v8f aR0 = {}, aR1 = {}, aW0 = {}, aW1 = {};
    v8f aK0 = {}, aK1 = {}, aV0 = {}, aV1 = {};
    for (int k0 = 0; k0 < K_; k0 += 32) {
        v16bf a0 = load_A(Xb, m0,      k0, lane);
        v16bf a1 = load_A(Xb, m0 + 16, k0, lane);
        v16bf bR = load_B(Wrb, n0, k0, lane);
        v16bf bW = load_B(Wwb, n0, k0, lane);
        v16bf bK = load_B(Wkb, n0, k0, lane);
        v16bf bV = load_B(Wvb, n0, k0, lane);
        aR0 = __builtin_amdgcn_wmma_f32_16x16x32_bf16(false, a0, false, bR, (short)0, aR0, false, false);
        aR1 = __builtin_amdgcn_wmma_f32_16x16x32_bf16(false, a1, false, bR, (short)0, aR1, false, false);
        aW0 = __builtin_amdgcn_wmma_f32_16x16x32_bf16(false, a0, false, bW, (short)0, aW0, false, false);
        aW1 = __builtin_amdgcn_wmma_f32_16x16x32_bf16(false, a1, false, bW, (short)0, aW1, false, false);
        aK0 = __builtin_amdgcn_wmma_f32_16x16x32_bf16(false, a0, false, bK, (short)0, aK0, false, false);
        aK1 = __builtin_amdgcn_wmma_f32_16x16x32_bf16(false, a1, false, bK, (short)0, aK1, false, false);
        aV0 = __builtin_amdgcn_wmma_f32_16x16x32_bf16(false, a0, false, bV, (short)0, aV0, false, false);
        aV1 = __builtin_amdgcn_wmma_f32_16x16x32_bf16(false, a1, false, bV, (short)0, aV1, false, false);
    }
    proj_epilogue(m0,      n0, lane, aR0, aW0, aK0, aV0, r_out, ewk_out, v_out);
    proj_epilogue(m0 + 16, n0, lane, aR1, aW1, aK1, aV1, r_out, ewk_out, v_out);
}

// ------------------------------------------------ scan phase A: chunk locals
__global__ __launch_bounds__(256) void scan_chunk_kernel(
    const float* __restrict__ ewk, const bf16* __restrict__ vbuf,
    const float* __restrict__ td,
    float* __restrict__ csNum, float* __restrict__ csDen)
{
    int tid = blockIdx.x * blockDim.x + threadIdx.x;    // B_*D_*NC threads
    int d  = tid % D_;
    int bc = tid / D_;
    int b  = bc % B_;
    int c  = bc / B_;
    float decay = __expf(td[d]);
    float num = 0.f, den = 0.f;
    size_t base = ((size_t)b * S_ + (size_t)c * CL) * D_ + d;
    for (int i = 0; i < CL; ++i) {
        float e  = ewk[base];
        float vv = (float)vbuf[base];
        num = decay * num + e * vv;
        den = decay * den + e;
        base += D_;
    }
    size_t o = ((size_t)c * B_ + b) * D_ + d;
    csNum[o] = num; csDen[o] = den;
}

// ------------------------------- scan phase B: serial prefix + final_state
__global__ __launch_bounds__(256) void scan_prefix_kernel(
    const float* __restrict__ csNum, const float* __restrict__ csDen,
    const float* __restrict__ td,
    float* __restrict__ caNum, float* __restrict__ caDen,
    float* __restrict__ fstate)   // d_out + M_*N_, layout [B][2][D]
{
    int tid = blockIdx.x * blockDim.x + threadIdx.x;    // B_*D_ threads
    int d = tid % D_, b = tid / D_;
    float dL = __expf(td[d] * (float)CL);               // decay^CL
    float num = 0.f, den = 0.f;
    for (int c = 0; c < NC; ++c) {
        size_t o = ((size_t)c * B_ + b) * D_ + d;
        caNum[o] = num; caDen[o] = den;                 // state entering chunk c
        num = dL * num + csNum[o];
        den = dL * den + csDen[o];
    }
    fstate[(size_t)b * 2 * D_ + d]      = num;
    fstate[(size_t)b * 2 * D_ + D_ + d] = den;
}

// --------------------------------------- scan phase C: apply carry, emit y
__global__ __launch_bounds__(256) void scan_apply_kernel(
    const float* __restrict__ ewk, const bf16* __restrict__ vbuf,
    const bf16* __restrict__ rbuf, const float* __restrict__ td,
    const float* __restrict__ caNum, const float* __restrict__ caDen,
    bf16* __restrict__ ybuf)
{
    int tid = blockIdx.x * blockDim.x + threadIdx.x;    // B_*D_*NC threads
    int d  = tid % D_;
    int bc = tid / D_;
    int b  = bc % B_;
    int c  = bc / B_;
    float decay = __expf(td[d]);
    size_t o = ((size_t)c * B_ + b) * D_ + d;
    float num = caNum[o], den = caDen[o];
    size_t base = ((size_t)b * S_ + (size_t)c * CL) * D_ + d;
    for (int i = 0; i < CL; ++i) {
        float e  = ewk[base];
        float vv = (float)vbuf[base];
        num = decay * num + e * vv;
        den = decay * den + e;
        float wkv = num / (den + 1e-8f);
        ybuf[base] = (bf16)((float)rbuf[base] * wkv);
        base += D_;
    }
}

// ------------------------------- output projection y @ Wo^T, 64x32 per wave
__global__ __launch_bounds__(256) void out_gemm_kernel(
    const bf16* __restrict__ Yb, const bf16* __restrict__ Wob,
    float* __restrict__ out)
{
    int lane = threadIdx.x & 31;
    int wave = blockIdx.x * (blockDim.x >> 5) + (threadIdx.x >> 5);
    int tn2 = wave & (N_ / 32 - 1);       // 0..63  (pair of N-tiles)
    int tq  = wave / (N_ / 32);           // 0..255 (quad of M-tiles)
    if (tq >= M_ / 64) return;            // wave-uniform
    int m0 = tq * 64, n0 = tn2 * 32;

    v8f acc[4][2] = {};
    for (int k0 = 0; k0 < K_; k0 += 32) {
        v16bf a[4];
#pragma unroll
        for (int i = 0; i < 4; ++i) a[i] = load_A(Yb, m0 + 16 * i, k0, lane);
        v16bf b0 = load_B(Wob, n0,      k0, lane);
        v16bf b1 = load_B(Wob, n0 + 16, k0, lane);
#pragma unroll
        for (int i = 0; i < 4; ++i) {
            acc[i][0] = __builtin_amdgcn_wmma_f32_16x16x32_bf16(false, a[i], false, b0, (short)0, acc[i][0], false, false);
            acc[i][1] = __builtin_amdgcn_wmma_f32_16x16x32_bf16(false, a[i], false, b1, (short)0, acc[i][1], false, false);
        }
    }
    int hf = lane >> 4, l = lane & 15;
#pragma unroll
    for (int i = 0; i < 4; ++i)
#pragma unroll
        for (int j = 0; j < 2; ++j)
#pragma unroll
            for (int e = 0; e < 8; ++e)
                out[(size_t)(m0 + 16 * i + e + 8 * hf) * N_ + (n0 + 16 * j + l)] = acc[i][j][e];
}

// ------------------------------------------------------------------ launcher
extern "C" void kernel_launch(void* const* d_in, const int* in_sizes, int n_in,
                              void* d_out, int out_size, void* d_ws, size_t ws_size,
                              hipStream_t stream) {
    (void)in_sizes; (void)n_in; (void)out_size; (void)ws_size;
    const float* x  = (const float*)d_in[0];
    const float* Wr = (const float*)d_in[1];
    const float* Ww = (const float*)d_in[2];
    const float* Wk = (const float*)d_in[3];
    const float* Wv = (const float*)d_in[4];
    const float* Wo = (const float*)d_in[5];
    const float* td = (const float*)d_in[6];
    float* out = (float*)d_out;

    char* p = (char*)d_ws;
    const size_t szX   = (size_t)M_ * K_ * sizeof(bf16);   // 64 MB
    const size_t szW   = (size_t)N_ * K_ * sizeof(bf16);   //  8 MB each
    const size_t szMN2 = (size_t)M_ * N_ * sizeof(bf16);   // 64 MB
    const size_t szMN4 = (size_t)M_ * N_ * sizeof(float);  // 128 MB
    const size_t szCs  = (size_t)NC * B_ * D_ * sizeof(float); // 1 MB

    bf16* xb  = (bf16*)p;  p += szX;
    bf16* wrb = (bf16*)p;  p += szW;
    bf16* wwb = (bf16*)p;  p += szW;
    bf16* wkb = (bf16*)p;  p += szW;
    bf16* wvb = (bf16*)p;  p += szW;
    bf16* wob = (bf16*)p;  p += szW;
    bf16* rb  = (bf16*)p;  p += szMN2;
    bf16* vb  = (bf16*)p;  p += szMN2;
    float* eb = (float*)p; p += szMN4;
    bf16* yb  = (bf16*)p;  p += szMN2;
    float* csN = (float*)p; p += szCs;
    float* csD = (float*)p; p += szCs;
    float* caN = (float*)p; p += szCs;
    float* caD = (float*)p; p += szCs;

    // bf16 conversions
    cvt_kernel<<<(M_ * K_ + 255) / 256, 256, 0, stream>>>(x,  xb,  M_ * K_);
    cvt_kernel<<<(N_ * K_ + 255) / 256, 256, 0, stream>>>(Wr, wrb, N_ * K_);
    cvt_kernel<<<(N_ * K_ + 255) / 256, 256, 0, stream>>>(Ww, wwb, N_ * K_);
    cvt_kernel<<<(N_ * K_ + 255) / 256, 256, 0, stream>>>(Wk, wkb, N_ * K_);
    cvt_kernel<<<(N_ * K_ + 255) / 256, 256, 0, stream>>>(Wv, wvb, N_ * K_);
    cvt_kernel<<<(N_ * K_ + 255) / 256, 256, 0, stream>>>(Wo, wob, N_ * K_);

    // fused 4-way projection GEMM (WMMA), 32x16 per wave, 8 waves/block
    const int proj_waves = (M_ / 32) * (N_ / 16);          // 65536
    proj_kernel<<<proj_waves / 8, 256, 0, stream>>>(xb, wrb, wwb, wkb, wvb, rb, eb, vb);

    // chunked linear-recurrence scan
    scan_chunk_kernel<<<B_ * D_ * NC / 256, 256, 0, stream>>>(eb, vb, td, csN, csD);
    scan_prefix_kernel<<<B_ * D_ / 256, 256, 0, stream>>>(csN, csD, td, caN, caD,
                                                          out + (size_t)M_ * N_);
    scan_apply_kernel<<<B_ * D_ * NC / 256, 256, 0, stream>>>(eb, vb, rb, td, caN, caD, yb);

    // output projection (WMMA), 64x32 per wave -> d_out
    const int og_waves = (M_ / 64) * (N_ / 32);            // 16384
    out_gemm_kernel<<<og_waves / 8, 256, 0, stream>>>(yb, wob, out);
}
